// PointNet2MSGExport_22196390986402
// MI455X (gfx1250) — matrix operand order repaired and checked by reading, not compiled, not run on hardware
//
#include <hip/hip_runtime.h>
#include <hip/hip_bf16.h>
#include <float.h>

#define B_ 2
#define N_ 8192

typedef __attribute__((ext_vector_type(16))) _Float16 v16h;
typedef __attribute__((ext_vector_type(8)))  float    v8f;

#define CDIV(a,b) (((a)+(b)-1)/(b))

// ---------------------------------------------------------------------------
// (B, C, N) -> (B, N, C)
__global__ void k_transpose_feats(const float* __restrict__ in, float* __restrict__ out,
                                  int n, int c) {
    int t = blockIdx.x * blockDim.x + threadIdx.x;
    int total = B_ * n * c;
    if (t >= total) return;
    int ni = t % n; int rest = t / n;
    int ci = rest % c; int b = rest / c;
    out[((size_t)b * n + ni) * c + ci] = in[t];
}

// ---------------------------------------------------------------------------
// Farthest point sampling: one block per batch, block-wide argmax reduction.
__global__ __launch_bounds__(256)
void k_fps(const float* __restrict__ xyz, int n, int npoint,
           int* __restrict__ out_idx, float* __restrict__ dist_ws) {
    int b = blockIdx.x;
    const float* px = xyz + (size_t)b * n * 3;
    float* dist = dist_ws + (size_t)b * n;
    __shared__ float sval[256];
    __shared__ int   sidx[256];
    int tid = threadIdx.x;
    for (int i = tid; i < n; i += 256) dist[i] = 1e10f;
    __syncthreads();
    int far = 0;
    for (int it = 0; it < npoint; ++it) {
        if (tid == 0) out_idx[b * npoint + it] = far;
        float cx = px[far * 3 + 0], cy = px[far * 3 + 1], cz = px[far * 3 + 2];
        float best = -1.0f; int bi = 0;
        for (int i = tid; i < n; i += 256) {
            float dx = px[i * 3 + 0] - cx;
            float dy = px[i * 3 + 1] - cy;
            float dz = px[i * 3 + 2] - cz;
            float d = dx * dx + dy * dy + dz * dz;
            float dm = dist[i];
            dm = dm < d ? dm : d;
            dist[i] = dm;
            if (dm > best) { best = dm; bi = i; }
        }
        sval[tid] = best; sidx[tid] = bi;
        __syncthreads();
        for (int s = 128; s > 0; s >>= 1) {
            if (tid < s) {
                if (sval[tid + s] > sval[tid] ||
                    (sval[tid + s] == sval[tid] && sidx[tid + s] < sidx[tid])) {
                    sval[tid] = sval[tid + s];
                    sidx[tid] = sidx[tid + s];
                }
            }
            __syncthreads();
        }
        far = sidx[0];
        __syncthreads();
    }
}

// ---------------------------------------------------------------------------
__global__ void k_gather_xyz(const float* __restrict__ xyz, const int* __restrict__ idx,
                             float* __restrict__ out, int n, int npoint) {
    int t = blockIdx.x * blockDim.x + threadIdx.x;
    if (t >= B_ * npoint) return;
    int b = t / npoint;
    int id = idx[t];
    const float* p = xyz + ((size_t)b * n + id) * 3;
    out[t * 3 + 0] = p[0];
    out[t * 3 + 1] = p[1];
    out[t * 3 + 2] = p[2];
}

// ---------------------------------------------------------------------------
// Ball query: first nsample indices within radius, padded with first hit;
// if none found, pad with n-1 (matches reference top_k construction).
__global__ void k_ball(const float* __restrict__ xyz, const float* __restrict__ new_xyz,
                       int n, int npoint, float r2, int nsample, int* __restrict__ out) {
    int t = blockIdx.x * blockDim.x + threadIdx.x;
    if (t >= B_ * npoint) return;
    int b = t / npoint;
    const float* nx = new_xyz + (size_t)t * 3;
    float cx = nx[0], cy = nx[1], cz = nx[2];
    int* o = out + (size_t)t * nsample;
    const float* px = xyz + (size_t)b * n * 3;
    int cnt = 0, first = n - 1;
    for (int i = 0; i < n && cnt < nsample; ++i) {
        float dx = px[i * 3 + 0] - cx;
        float dy = px[i * 3 + 1] - cy;
        float dz = px[i * 3 + 2] - cz;
        float d2 = dx * dx + dy * dy + dz * dz;
        if (d2 < r2) {
            if (cnt == 0) first = i;
            o[cnt++] = i;
        }
    }
    if (cnt == 0)
        for (int s = 0; s < nsample; ++s) o[s] = n - 1;
    else
        for (int s = cnt; s < nsample; ++s) o[s] = first;
}

// ---------------------------------------------------------------------------
// Build grouped tensor g[row][c], row = (b*npoint+p)*nsample+s.
__global__ void k_group(const float* __restrict__ xyz, const float* __restrict__ feats,
                        const float* __restrict__ new_xyz, const int* __restrict__ idx,
                        float* __restrict__ g, int n, int npoint, int nsample, int cin) {
    int C = cin + 3;
    size_t total = (size_t)B_ * npoint * nsample * C;
    size_t t = (size_t)blockIdx.x * blockDim.x + threadIdx.x;
    if (t >= total) return;
    int c = (int)(t % C); size_t rest = t / C;
    int s = (int)(rest % nsample); rest /= nsample;
    int p = (int)(rest % npoint);
    int b = (int)(rest / npoint);
    int id = idx[((size_t)(b * npoint) + p) * nsample + s];
    float v;
    if (c < 3)
        v = xyz[((size_t)b * n + id) * 3 + c] - new_xyz[((size_t)b * npoint + p) * 3 + c];
    else
        v = feats[((size_t)b * n + id) * cin + (c - 3)];
    g[t] = v;
}

// ---------------------------------------------------------------------------
// WMMA GEMM + BN-scale + ReLU:  Y[r][o] = relu(gamma[o]*sum_c X[r][c]*W[o][c] + beta[o])
// One wave per 16x(NT*16) tile of Y; A fragment reused across NT B tiles.
// Branch-free main K loop; K tail hoisted out; partial column tiles zero-filled.
template <int NT>
__global__ __launch_bounds__(32)
void k_gemm_bn_relu(const float* __restrict__ X, const float* __restrict__ W,
                    const float* __restrict__ gamma, const float* __restrict__ beta,
                    float* __restrict__ Y, int R, int Cin, int Cout) {
    int lane = threadIdx.x;            // 0..31, EXEC all-1s in uniform regions
    int r0 = blockIdx.x * 16;
    int o0 = blockIdx.y * (NT * 16);
    int m  = lane & 15;
    int hi = lane >> 4;
    int ka = hi * 8;                   // A: lanes 16-31 hold K 8..15 / 24..31
    int kb = hi * 8;                   // B: lanes 16-31 hold K 8..15 / 24..31

    int ar = r0 + m; if (ar >= R) ar = R - 1;           // rows here are %16==0
    const float* __restrict__ arow = X + (size_t)ar * Cin;

    bool oOK[NT];
    const float* __restrict__ wrow[NT];
    #pragma unroll
    for (int j = 0; j < NT; ++j) {
        int oc = o0 + j * 16 + m;
        oOK[j]  = oc < Cout;
        wrow[j] = W + (size_t)(oOK[j] ? oc : 0) * Cin;
    }

    v8f acc[NT];
    #pragma unroll
    for (int j = 0; j < NT; ++j) acc[j] = v8f{};

    int kfull = Cin & ~31;             // main-loop bound (multiple of 32)

    // ---- branch-free main K loop (full 32-wide steps)
    for (int k0 = 0; k0 < kfull; k0 += 32) {
        __builtin_prefetch(arow + k0 + 32 + ka, 0, 1);
        v16h a;
        {
            float af[16];
            #pragma unroll
            for (int i = 0; i < 8; ++i) af[i]     = arow[k0 + ka + i];
            #pragma unroll
            for (int i = 0; i < 8; ++i) af[8 + i] = arow[k0 + 16 + ka + i];
            #pragma unroll
            for (int i = 0; i < 16; ++i) a[i] = (_Float16)af[i];
        }
        #pragma unroll
        for (int j = 0; j < NT; ++j) {
            v16h b;
            if (oOK[j]) {
                float bf[16];
                #pragma unroll
                for (int i = 0; i < 16; ++i) bf[i] = wrow[j][k0 + kb + (i < 8 ? i : 8 + i)];
                #pragma unroll
                for (int i = 0; i < 16; ++i) b[i] = (_Float16)bf[i];
            } else {
                #pragma unroll
                for (int i = 0; i < 16; ++i) b[i] = (_Float16)0.0f;
            }
            acc[j] = __builtin_amdgcn_wmma_f32_16x16x32_f16(
                /*neg_a=*/false, a, /*neg_b=*/false, b,
                /*c_mod=*/(short)0, acc[j], /*reuse_a=*/false, /*reuse_b=*/false);
        }
    }

    // ---- hoisted K tail (Cin % 32 != 0), zero-padded
    if (kfull < Cin) {
        int k0 = kfull;
        v16h a;
        #pragma unroll
        for (int i = 0; i < 8; ++i) {
            int k = k0 + ka + i;
            a[i] = (k < Cin) ? (_Float16)arow[k] : (_Float16)0.0f;
        }
        #pragma unroll
        for (int i = 0; i < 8; ++i) {
            int k = k0 + 16 + ka + i;
            a[8 + i] = (k < Cin) ? (_Float16)arow[k] : (_Float16)0.0f;
        }
        #pragma unroll
        for (int j = 0; j < NT; ++j) {
            v16h b;
            #pragma unroll
            for (int i = 0; i < 16; ++i) {
                int k = k0 + kb + (i < 8 ? i : 8 + i);
                b[i] = (oOK[j] && k < Cin) ? (_Float16)wrow[j][k] : (_Float16)0.0f;
            }
            acc[j] = __builtin_amdgcn_wmma_f32_16x16x32_f16(
                false, a, false, b, (short)0, acc[j], false, false);
        }
    }

    // ---- Epilogue: D VGPR v -> row v + 8*hi, col lane&15.
    #pragma unroll
    for (int j = 0; j < NT; ++j) {
        int o = o0 + j * 16 + m;
        float g  = oOK[j] ? gamma[o] : 0.0f;
        float bt = oOK[j] ? beta[o]  : 0.0f;
        #pragma unroll
        for (int v = 0; v < 8; ++v) {
            int r = r0 + v + 8 * hi;
            if (r < R && oOK[j]) {
                float y = acc[j][v] * g + bt;
                Y[(size_t)r * Cout + o] = y > 0.0f ? y : 0.0f;
            }
        }
    }
}

// Host-side dispatch picking the column-tile blocking factor.
static void launch_gemm(const float* X, const float* W, const float* ga, const float* be,
                        float* Y, int R, int Cin, int Cout, hipStream_t stream) {
    int tiles = CDIV(Cout, 16);
    int nt;
    if (tiles % 4 == 0 || tiles > 8) nt = 4;
    else if (tiles % 3 == 0)         nt = 3;
    else if (tiles % 2 == 0)         nt = 2;
    else                             nt = (tiles >= 4) ? 4 : 1;
    dim3 grid((unsigned)CDIV(R, 16), (unsigned)CDIV(Cout, nt * 16));
    switch (nt) {
    case 1: k_gemm_bn_relu<1><<<grid, 32, 0, stream>>>(X, W, ga, be, Y, R, Cin, Cout); break;
    case 2: k_gemm_bn_relu<2><<<grid, 32, 0, stream>>>(X, W, ga, be, Y, R, Cin, Cout); break;
    case 3: k_gemm_bn_relu<3><<<grid, 32, 0, stream>>>(X, W, ga, be, Y, R, Cin, Cout); break;
    default: k_gemm_bn_relu<4><<<grid, 32, 0, stream>>>(X, W, ga, be, Y, R, Cin, Cout); break;
    }
}

// ---------------------------------------------------------------------------
// Max over nsample; scatter into concatenated per-level feature buffer.
__global__ void k_maxpool(const float* __restrict__ Yin, float* __restrict__ out,
                          int npoint, int nsample, int cout, int cstride, int coffset) {
    size_t t = (size_t)blockIdx.x * blockDim.x + threadIdx.x;
    size_t total = (size_t)B_ * npoint * cout;
    if (t >= total) return;
    int c = (int)(t % cout); size_t rest = t / cout;
    int p = (int)(rest % npoint);
    int b = (int)(rest / npoint);
    const float* y = Yin + ((size_t)(b * npoint + p) * nsample) * cout + c;
    float m = -FLT_MAX;
    for (int s = 0; s < nsample; ++s) {
        float v = y[(size_t)s * cout];
        m = v > m ? v : m;
    }
    out[((size_t)b * npoint + p) * cstride + coffset + c] = m;
}

// ---------------------------------------------------------------------------
// FP: 3-NN inverse-distance interpolation + concat(interp, unk_feats).
__global__ void k_fp_interp_concat(const float* __restrict__ unk_xyz,
                                   const float* __restrict__ kn_xyz,
                                   const float* __restrict__ unk_feats,
                                   const float* __restrict__ kn_feats,
                                   float* __restrict__ out,
                                   int n_unk, int n_kn, int pre, int skipc) {
    int t = blockIdx.x * blockDim.x + threadIdx.x;
    if (t >= B_ * n_unk) return;
    int b = t / n_unk;
    float px = unk_xyz[(size_t)t * 3 + 0];
    float py = unk_xyz[(size_t)t * 3 + 1];
    float pz = unk_xyz[(size_t)t * 3 + 2];
    const float* kx = kn_xyz + (size_t)b * n_kn * 3;
    float bd0 = 1e30f, bd1 = 1e30f, bd2 = 1e30f;
    int   bi0 = 0,     bi1 = 0,     bi2 = 0;
    for (int j = 0; j < n_kn; ++j) {
        float dx = kx[j * 3 + 0] - px;
        float dy = kx[j * 3 + 1] - py;
        float dz = kx[j * 3 + 2] - pz;
        float d2 = dx * dx + dy * dy + dz * dz;
        if (d2 < bd0)      { bd2=bd1; bi2=bi1; bd1=bd0; bi1=bi0; bd0=d2; bi0=j; }
        else if (d2 < bd1) { bd2=bd1; bi2=bi1; bd1=d2; bi1=j; }
        else if (d2 < bd2) { bd2=d2; bi2=j; }
    }
    float w0 = 1.0f / (bd0 + 1e-8f);
    float w1 = 1.0f / (bd1 + 1e-8f);
    float w2 = 1.0f / (bd2 + 1e-8f);
    float wsum = w0 + w1 + w2;
    w0 /= wsum; w1 /= wsum; w2 /= wsum;
    const float* f0 = kn_feats + ((size_t)b * n_kn + bi0) * pre;
    const float* f1 = kn_feats + ((size_t)b * n_kn + bi1) * pre;
    const float* f2 = kn_feats + ((size_t)b * n_kn + bi2) * pre;
    float* o = out + (size_t)t * (pre + skipc);
    for (int c = 0; c < pre; ++c)
        o[c] = w0 * f0[c] + w1 * f1[c] + w2 * f2[c];
    const float* uf = unk_feats + (size_t)t * skipc;
    for (int c = 0; c < skipc; ++c)
        o[pre + c] = uf[c];
}

// ---------------------------------------------------------------------------
extern "C" void kernel_launch(void* const* d_in, const int* in_sizes, int n_in,
                              void* d_out, int out_size, void* d_ws, size_t ws_size,
                              hipStream_t stream) {
    (void)in_sizes; (void)n_in; (void)out_size; (void)ws_size;

    static const int   NPOINTS_[4]      = {2048, 512, 128, 32};
    static const float RADII_[4][2]     = {{0.1f,0.5f},{0.5f,1.0f},{1.0f,2.0f},{2.0f,4.0f}};
    static const int   NSAMP_[4][2]     = {{16,32},{16,32},{16,32},{16,32}};
    static const int   MLPS_[4][2][3]   = {{{16,16,32},{32,32,64}},
                                           {{64,64,128},{64,96,128}},
                                           {{128,196,256},{128,196,256}},
                                           {{256,256,512},{256,384,512}}};
    static const int   FP_MLPS_[4][2]   = {{128,128},{256,256},{512,512},{512,512}};
    static const int   SKIP_[5]         = {1, 96, 256, 512, 1024};
    static const int   CNT_[5]          = {N_, 2048, 512, 128, 32};

    const float* xyz0    = (const float*)d_in[0];
    const float* feat_in = (const float*)d_in[1];
    float* ws = (float*)d_ws;

    // ---- static workspace layout (float units); total ~18.6M floats (~75 MB)
    size_t off = 0;
    auto A = [&](size_t nf) { size_t o = off; off += nf; return o; };
    float* lf[5];
    lf[0] = ws + A((size_t)B_ * N_ * 1);
    lf[1] = ws + A((size_t)B_ * 2048 * 96);
    lf[2] = ws + A((size_t)B_ * 512 * 256);
    lf[3] = ws + A((size_t)B_ * 128 * 512);
    lf[4] = ws + A((size_t)B_ * 32 * 1024);
    float* nxyz[4];
    nxyz[0] = ws + A((size_t)B_ * 2048 * 3);
    nxyz[1] = ws + A((size_t)B_ * 512 * 3);
    nxyz[2] = ws + A((size_t)B_ * 128 * 3);
    nxyz[3] = ws + A((size_t)B_ * 32 * 3);
    float* fpbuf3 = ws + A((size_t)B_ * 128 * 512);
    float* fpbuf2 = ws + A((size_t)B_ * 512 * 512);
    float* fpbuf1 = ws + A((size_t)B_ * 2048 * 256);
    float* fps_dist = ws + A((size_t)B_ * N_);
    int*   fps_idx  = (int*)(ws + A((size_t)B_ * 2048));
    int*   ball_idx = (int*)(ws + A((size_t)B_ * 2048 * 32));
    float* g_buf = ws + A((size_t)3244032);   // max grouped tensor (level1 scale1)
    float* act_a = ws + A((size_t)8388608);   // max activation / FP concat
    float* act_b = ws + A((size_t)4194304);

    // ---- features (B,1,N) -> (B,N,1)
    k_transpose_feats<<<CDIV(B_ * N_ * 1, 256), 256, 0, stream>>>(feat_in, lf[0], N_, 1);

    // ---- Set-abstraction levels
    const float* cur_xyz = xyz0;
    int ncur = N_;
    const float* fcur = lf[0];
    int pidx = 2;   // d_in cursor into SA params
    for (int k = 0; k < 4; ++k) {
        int npoint = NPOINTS_[k];
        k_fps<<<B_, 256, 0, stream>>>(cur_xyz, ncur, npoint, fps_idx, fps_dist);
        k_gather_xyz<<<CDIV(B_ * npoint, 128), 128, 0, stream>>>(cur_xyz, fps_idx,
                                                                 nxyz[k], ncur, npoint);
        int cin = SKIP_[k];
        int coff = 0;
        for (int s = 0; s < 2; ++s) {
            int ns = NSAMP_[k][s];
            float r = RADII_[k][s];
            k_ball<<<CDIV(B_ * npoint, 64), 64, 0, stream>>>(cur_xyz, nxyz[k], ncur,
                                                             npoint, r * r, ns, ball_idx);
            int C = cin + 3;
            size_t rows = (size_t)B_ * npoint * ns;
            size_t gel = rows * C;
            k_group<<<(unsigned)CDIV(gel, 256), 256, 0, stream>>>(cur_xyz, fcur, nxyz[k],
                                                                  ball_idx, g_buf,
                                                                  ncur, npoint, ns, cin);
            const float* X = g_buf;
            int ci = C;
            float* bufs[2] = {act_a, act_b};
            for (int l = 0; l < 3; ++l) {
                int co = MLPS_[k][s][l];
                const float* W  = (const float*)d_in[pidx + 0];
                const float* ga = (const float*)d_in[pidx + 1];
                const float* be = (const float*)d_in[pidx + 2];
                pidx += 3;
                float* Y = bufs[l & 1];
                launch_gemm(X, W, ga, be, Y, (int)rows, ci, co, stream);
                X = Y; ci = co;
            }
            k_maxpool<<<(unsigned)CDIV((size_t)B_ * npoint * ci, 256), 256, 0, stream>>>(
                X, lf[k + 1], npoint, ns, ci, SKIP_[k + 1], coff);
            coff += ci;
        }
        cur_xyz = nxyz[k];
        ncur = npoint;
        fcur = lf[k + 1];
    }

    // ---- Feature propagation (coarse -> fine), last level writes d_out
    const float* xyzl[5] = {xyz0, nxyz[0], nxyz[1], nxyz[2], nxyz[3]};
    float* fpdst[4] = {(float*)d_out, fpbuf1, fpbuf2, fpbuf3};
    const float* knf = lf[4];
    int knc = 1024;
    for (int i = 3; i >= 0; --i) {
        int nu = CNT_[i], nk = CNT_[i + 1];
        int pre = knc, skipc = SKIP_[i];
        int cc = pre + skipc;
        k_fp_interp_concat<<<CDIV(B_ * nu, 128), 128, 0, stream>>>(
            xyzl[i], xyzl[i + 1], lf[i], knf, act_a, nu, nk, pre, skipc);
        int base = 74 + i * 6;   // 2 + 72 SA arrays, then fp_params[i] = 2 layers x 3
        int rows = B_ * nu;
        int c0 = FP_MLPS_[i][0];
        {
            const float* W  = (const float*)d_in[base + 0];
            const float* ga = (const float*)d_in[base + 1];
            const float* be = (const float*)d_in[base + 2];
            launch_gemm(act_a, W, ga, be, act_b, rows, cc, c0, stream);
        }
        int c1 = FP_MLPS_[i][1];
        {
            const float* W  = (const float*)d_in[base + 3];
            const float* ga = (const float*)d_in[base + 4];
            const float* be = (const float*)d_in[base + 5];
            launch_gemm(act_b, W, ga, be, fpdst[i], rows, c0, c1, stream);
        }
        knf = fpdst[i];
        knc = c1;
    }
}
